// AttentionEncoderDecoderModel_85332410237552
// MI455X (gfx1250) — matrix operand order repaired and checked by reading, not compiled, not run on hardware
//
#include <hip/hip_runtime.h>
#include <math.h>

// ---------------- problem constants ----------------
#define BB 32
#define TT 64
#define TY 32
#define VV 50000
#define EE 256
#define HH2 512          // H
#define HHH 256          // H/2

typedef __bf16 bf16;
typedef __attribute__((ext_vector_type(16))) __bf16 v16bf;
typedef __attribute__((ext_vector_type(8)))  float  v8f;

// ---- CDNA5 async global->LDS copies (ASYNCcnt-tracked, no VGPR bounce) ----
__device__ __forceinline__ void async_g2l_b128(unsigned lds_off, const void* gaddr)
{
    asm volatile("global_load_async_to_lds_b128 %0, %1, off"
                 :: "v"(lds_off), "v"((unsigned long long)(size_t)gaddr)
                 : "memory");
}
__device__ __forceinline__ void async_g2l_b64(unsigned lds_off, const void* gaddr)
{
    asm volatile("global_load_async_to_lds_b64 %0, %1, off"
                 :: "v"(lds_off), "v"((unsigned long long)(size_t)gaddr)
                 : "memory");
}
__device__ __forceinline__ void wait_async0()
{
    asm volatile("s_wait_asynccnt 0x0" ::: "memory");
}

// =====================================================================
// WMMA GEMM:  C(MxN,f32) = A(MxK,bf16,rm) * B(KxN,bf16,rm) + bias(N)
// block = 256 threads = 8 waves; block tile 32(M) x 128(N); wave tile
// 16x32 (2 WMMAs sharing one A fragment); K-slab 32 staged via LDS with
// GLOBAL_LOAD_ASYNC_TO_LDS (ASYNCcnt) instead of a VGPR bounce.
// Requirements (all satisfied at call sites): K%32==0, M%32==0, N%8==0,
// N>=16, bias!=null.  N-edge tiles carry NO slow path: B loads are
// address-clamped to [0, N-16], so out-of-range LDS columns hold
// duplicated valid data that is computed on but never stored (the store
// has the only n<N guard).
// =====================================================================
#define TNB 128
__global__ __launch_bounds__(256)
void gemm_bf16_wmma(const bf16* __restrict__ A, const bf16* __restrict__ B,
                    const float* __restrict__ bias,
                    float* __restrict__ C,
                    int M, int N, int K)
{
    __shared__ bf16 As[32][40];        // 32x32 tile, +8 pad (80B row stride)
    __shared__ bf16 Bs[32][TNB + 8];   // 32x128 tile, +8 pad (272B row stride)

    const int tid  = threadIdx.x;
    const int lane = tid & 31;
    const int wave = tid >> 5;
    const int tm   = wave >> 2;              // 0..1  (M sub-tile)
    const int tn   = wave & 3;               // 0..3  (32-col slab per wave)
    const int mBase = blockIdx.y * 32;
    const int nBase = blockIdx.x * TNB;

    // cooperative fill assignments (address-clamped; see header comment)
    const int ar = tid >> 3;                 // A row 0..31
    const int ac = (tid & 7) << 2;           // A col, 4 bf16 per thread
    const int kr = tid >> 3;                 // B k-row 0..31
    const int nc = (tid & 7) << 4;           // B n-col, 16 bf16 per thread

    int bn = nBase + nc;        if (bn > N - 16) bn = N - 16;

    // loop-invariant LDS byte offsets (addrspace truncation = LDS address)
    const unsigned ldsA  = (unsigned)(size_t)&As[ar][ac];
    const unsigned ldsB0 = (unsigned)(size_t)&Bs[kr][nc];
    const unsigned ldsB1 = (unsigned)(size_t)&Bs[kr][nc + 8];

    // incremented global pointers (no per-iteration 64-bit mul)
    const bf16* aPtr = A + (size_t)(mBase + ar) * K + ac;
    const bf16* bPtr = B + (size_t)kr * N + bn;
    const size_t bStep = (size_t)32 * N;

    // fragment addressing (documented CDNA5 WMMA layouts)
    const int arow = tm * 16 + (lane & 15);
    const int kb   = (lane >> 4) << 3;       // 0 or 8

    v8f acc0 = {}, acc1 = {};

    for (int k0 = 0; k0 < K; k0 += 32) {
        async_g2l_b64 (ldsA,  aPtr);
        async_g2l_b128(ldsB0, bPtr);
        async_g2l_b128(ldsB1, bPtr + 8);
        // unconditional speculative prefetch of the next K slab of B;
        // a tail over-prefetch is dropped on translation failure (ISA §10.5)
        __builtin_prefetch(bPtr + bStep, 0, 1);
        aPtr += 32;
        bPtr += bStep;

        wait_async0();        // own async stores to LDS complete
        __syncthreads();      // everyone's tile data visible

        v16bf af, b0, b1;
        #pragma unroll
        for (int v = 0; v < 4; ++v) {
            af[2 * v]         = As[arow][kb + 2 * v];
            af[2 * v + 1]     = As[arow][kb + 2 * v + 1];
            af[8 + 2 * v]     = As[arow][16 + kb + 2 * v];
            af[8 + 2 * v + 1] = As[arow][16 + kb + 2 * v + 1];
        }
        const int bcol = tn * 32 + (lane & 15);
        #pragma unroll
        for (int v = 0; v < 8; ++v) {
            b0[v]     = Bs[kb + v][bcol];
            b0[8 + v] = Bs[16 + kb + v][bcol];
            b1[v]     = Bs[kb + v][bcol + 16];
            b1[8 + v] = Bs[16 + kb + v][bcol + 16];
        }

        acc0 = __builtin_amdgcn_wmma_f32_16x16x32_bf16(
                   false, af, false, b0, (short)0, acc0, false, false);
        acc1 = __builtin_amdgcn_wmma_f32_16x16x32_bf16(
                   false, af, false, b1, (short)0, acc1, false, false);
        __syncthreads();
    }

    // ---- store: C/D layout: VGPR r -> M = r + 8*(lane>=16), N = lane&15 ----
    // M%32==0 guaranteed; only the N edge needs a guard.
    const int m0 = mBase + tm * 16 + ((lane >> 4) << 3);
    #pragma unroll
    for (int half = 0; half < 2; ++half) {
        const int n = nBase + tn * 32 + half * 16 + (lane & 15);
        if (n < N) {
            const float bv = bias[n];
            const v8f acc = half ? acc1 : acc0;
            float* cp = C + (size_t)m0 * N + n;
            #pragma unroll
            for (int r = 0; r < 8; ++r) {
                cp[(size_t)r * N] = acc[r] + bv;
            }
        }
    }
}

// =====================================================================
// Weight prep: out[(K1+K2) x N] (bf16) = transpose of [W1 | W2] where
// W1 is N x K1 and W2 is N x K2 (row-major f32). W2 may be null (K2=0).
// =====================================================================
__global__ void pack_cat2_T(const float* __restrict__ W1, int K1,
                            const float* __restrict__ W2, int K2,
                            bf16* __restrict__ out, int N)
{
    const long total = (long)(K1 + K2) * N;
    long idx = (long)blockIdx.x * blockDim.x + threadIdx.x;
    if (idx >= total) return;
    const int k = (int)(idx / N);
    const int n = (int)(idx % N);
    const float v = (k < K1) ? W1[(size_t)n * K1 + k]
                             : W2[(size_t)n * K2 + (k - K1)];
    out[idx] = (bf16)v;
}

__global__ void vec_add2(const float* a, const float* b, float* o, int n)
{
    int i = blockIdx.x * blockDim.x + threadIdx.x;
    if (i < n) o[i] = a[i] + b[i];
}

__global__ void copy_f32(const float* s, float* d, int n)
{
    int i = blockIdx.x * blockDim.x + threadIdx.x;
    if (i < n) d[i] = s[i];
}

// gather: out[row, 0..E) = table[ids[row], 0..E)   (block per row, E threads)
__global__ void gather_embed(const int* __restrict__ ids,
                             const float* __restrict__ table,
                             float* __restrict__ out, int E)
{
    const int r = blockIdx.x;
    const int e = threadIdx.x;
    out[(size_t)r * E + e] = table[(size_t)ids[r] * E + e];
}

// length-aware reverse of token ids (one block per batch row, T threads)
__global__ void build_rev_ids(const int* __restrict__ x,
                              const int* __restrict__ xmask,
                              int* __restrict__ xb, int T)
{
    const int b = blockIdx.x, t = threadIdx.x;
    __shared__ int red[TT];
    red[t] = 1 - xmask[b * T + t];
    __syncthreads();
    for (int s = TT / 2; s > 0; s >>= 1) {
        if (t < s) red[t] += red[t + s];
        __syncthreads();
    }
    const int mask_len = red[0];
    xb[b * T + t] = 0;
    __syncthreads();
    int rev = T - 1 - t - mask_len;
    if (rev < 0) rev = 0;
    xb[b * T + rev] = x[b * T + t];
}

// out[row] = bf16([a | b | c]) ; any of b/c may be null
__global__ void concat3_cast(const float* __restrict__ a, long as_, int n1,
                             const float* __restrict__ b, long bs_, int n2,
                             const float* __restrict__ c, long cs_, int n3,
                             bf16* __restrict__ out)
{
    const int r = blockIdx.x;
    const int ncols = n1 + n2 + n3;
    for (int col = threadIdx.x; col < ncols; col += blockDim.x) {
        float v;
        if (col < n1)            v = a[(size_t)r * as_ + col];
        else if (col < n1 + n2)  v = b[(size_t)r * bs_ + (col - n1)];
        else                     v = c[(size_t)r * cs_ + (col - n1 - n2)];
        out[(size_t)r * ncols + col] = (bf16)v;
    }
}

__device__ __forceinline__ float sigm(float x) { return 1.0f / (1.0f + expf(-x)); }

// LSTM gate activation. gates: rows x 4Hd (order i,f,g,o). Updates h/c state
// in place and mirrors h into optional f32/bf16 time-series slices.
__global__ void lstm_act(const float* __restrict__ gates,
                         float* __restrict__ cState, float* __restrict__ hState,
                         bf16* __restrict__ hB,
                         float* __restrict__ hX,  long hXs,
                         bf16*  __restrict__ hXB, long hXBs,
                         float* __restrict__ cX,  long cXs,
                         int Hd, int rows)
{
    int idx = blockIdx.x * blockDim.x + threadIdx.x;
    if (idx >= rows * Hd) return;
    const int r = idx / Hd, h = idx % Hd;
    const float* g = gates + (size_t)r * 4 * Hd;
    const float ig = sigm(g[h]);
    const float fg = sigm(g[Hd + h]);
    const float gg = tanhf(g[2 * Hd + h]);
    const float og = sigm(g[3 * Hd + h]);
    const float c  = fg * cState[r * Hd + h] + ig * gg;
    const float hh = og * tanhf(c);
    cState[r * Hd + h] = c;
    hState[r * Hd + h] = hh;
    if (hB)  hB[r * Hd + h] = (bf16)hh;
    if (hX)  hX[(size_t)r * hXs + h] = hh;
    if (hXB) hXB[(size_t)r * hXBs + h] = (bf16)hh;
    if (cX)  cX[(size_t)r * cXs + h] = c;
}

// dst[b,h] = mean_t src[b,t,h]
__global__ void mean_T(const float* __restrict__ src, float* __restrict__ dst,
                       int T, int Hd)
{
    int idx = blockIdx.x * blockDim.x + threadIdx.x;
    const int b = idx / Hd, h = idx % Hd;
    float s = 0.0f;
    for (int t = 0; t < T; ++t) s += src[((size_t)b * T + t) * Hd + h];
    dst[idx] = s * (1.0f / T);
}

// pre_bf16[b,t,h] = bf16(shx[b,h] + hproj[b,t,h])
__global__ void add_bcast_cast(const float* __restrict__ shx,
                               const float* __restrict__ hproj,
                               bf16* __restrict__ out, int T, int Hd)
{
    long idx = (long)blockIdx.x * blockDim.x + threadIdx.x;
    const int b = (int)(idx / ((long)T * Hd));
    const int h = (int)(idx % Hd);
    out[idx] = (bf16)(shx[(size_t)b * Hd + h] + hproj[idx]);
}

// in-place softmax over the last dim (cols), one block per row
__global__ void softmax_rows(float* __restrict__ x, int cols)
{
    float* row = x + (size_t)blockIdx.x * cols;
    __shared__ float red[256];
    const int tid = threadIdx.x;
    float m = -3.4e38f;
    for (int c = tid; c < cols; c += 256) m = fmaxf(m, row[c]);
    red[tid] = m; __syncthreads();
    for (int s = 128; s > 0; s >>= 1) { if (tid < s) red[tid] = fmaxf(red[tid], red[tid + s]); __syncthreads(); }
    m = red[0]; __syncthreads();
    float sum = 0.0f;
    for (int c = tid; c < cols; c += 256) { float e = expf(row[c] - m); row[c] = e; sum += e; }
    red[tid] = sum; __syncthreads();
    for (int s = 128; s > 0; s >>= 1) { if (tid < s) red[tid] += red[tid + s]; __syncthreads(); }
    const float inv = 1.0f / red[0];
    for (int c = tid; c < cols; c += 256) row[c] *= inv;
}

// ctx[b,h] = sum_t hid[b,t,h] * att[b,t,h]
__global__ void ctx_reduce(const float* __restrict__ hid,
                           const float* __restrict__ att,
                           float* __restrict__ ctx, int T, int Hd)
{
    int idx = blockIdx.x * blockDim.x + threadIdx.x;
    const int b = idx / Hd, h = idx % Hd;
    float s = 0.0f;
    for (int t = 0; t < T; ++t) {
        const size_t o = ((size_t)b * T + t) * Hd + h;
        s += hid[o] * att[o];
    }
    ctx[idx] = s;
}

// in-place log_softmax over cols, one block per row (3 passes over V)
__global__ void log_softmax_rows(float* __restrict__ x, int cols)
{
    float* row = x + (size_t)blockIdx.x * cols;
    __shared__ float red[256];
    const int tid = threadIdx.x;
    float m = -3.4e38f;
    for (int c = tid; c < cols; c += 256) m = fmaxf(m, row[c]);
    red[tid] = m; __syncthreads();
    for (int s = 128; s > 0; s >>= 1) { if (tid < s) red[tid] = fmaxf(red[tid], red[tid + s]); __syncthreads(); }
    m = red[0]; __syncthreads();
    float sum = 0.0f;
    for (int c = tid; c < cols; c += 256) sum += expf(row[c] - m);
    red[tid] = sum; __syncthreads();
    for (int s = 128; s > 0; s >>= 1) { if (tid < s) red[tid] += red[tid + s]; __syncthreads(); }
    const float sub = m + logf(red[0]);
    for (int c = tid; c < cols; c += 256) row[c] -= sub;
}

// =====================================================================
// Host orchestration
// =====================================================================
static inline void launch_gemm(const bf16* A, const bf16* B, const float* bias,
                               float* C, int M, int N, int K, hipStream_t s)
{
    dim3 g((N + TNB - 1) / TNB, (M + 31) / 32);
    gemm_bf16_wmma<<<g, 256, 0, s>>>(A, B, bias, C, M, N, K);
}

extern "C" void kernel_launch(void* const* d_in, const int* in_sizes, int n_in,
                              void* d_out, int out_size, void* d_ws, size_t ws_size,
                              hipStream_t stream)
{
    (void)in_sizes; (void)n_in; (void)out_size; (void)ws_size;
    const int*   x      = (const int*)  d_in[0];
    const int*   xmask  = (const int*)  d_in[1];
    const int*   y      = (const int*)  d_in[2];
    const float* f_h0   = (const float*)d_in[3];
    const float* f_c0   = (const float*)d_in[4];
    const float* b_h0   = (const float*)d_in[5];
    const float* b_c0   = (const float*)d_in[6];
    const float* embedW = (const float*)d_in[7];
    const float* fWih = (const float*)d_in[8],  *fWhh = (const float*)d_in[9];
    const float* fbih = (const float*)d_in[10], *fbhh = (const float*)d_in[11];
    const float* bWih = (const float*)d_in[12], *bWhh = (const float*)d_in[13];
    const float* bbih = (const float*)d_in[14], *bbhh = (const float*)d_in[15];
    const float* dWih = (const float*)d_in[16], *dWhh = (const float*)d_in[17];
    const float* dbih = (const float*)d_in[18], *dbhh = (const float*)d_in[19];
    const float* sW = (const float*)d_in[20], *sb = (const float*)d_in[21];
    const float* hW = (const float*)d_in[22], *hb = (const float*)d_in[23];
    const float* aW = (const float*)d_in[24], *ab = (const float*)d_in[25];
    const float* outW = (const float*)d_in[26], *outb = (const float*)d_in[27];

    float* out = (float*)d_out;
    float* logits = out;                                      // (B*TY) x V
    float* out_h  = out + (size_t)BB * TY * VV;               // (B,TY,H)

    // ---- workspace bump allocator ----
    char* p = (char*)d_ws;
    auto carve = [&](size_t bytes) -> void* {
        void* r = (void*)p;
        p += (bytes + 255) & ~(size_t)255;
        return r;
    };
    bf16* fWcat = (bf16*)carve((size_t)(EE + HHH) * 4 * HHH * 2);   // 512x1024
    bf16* bWcat = (bf16*)carve((size_t)(EE + HHH) * 4 * HHH * 2);
    bf16* dWcat = (bf16*)carve((size_t)(EE + 2 * HH2) * 4 * HH2 * 2); // 1280x2048
    bf16* sWT   = (bf16*)carve((size_t)HH2 * HH2 * 2);
    bf16* hWT   = (bf16*)carve((size_t)HH2 * HH2 * 2);
    bf16* aWT   = (bf16*)carve((size_t)HH2 * HH2 * 2);
    bf16* outWT = (bf16*)carve((size_t)HH2 * VV * 2);               // 512x50000
    float* fbias = (float*)carve(4 * HHH * 4);
    float* bbias = (float*)carve(4 * HHH * 4);
    float* dbias = (float*)carve(4 * HH2 * 4);
    float* x_emb  = (float*)carve((size_t)BB * TT * EE * 4);
    float* xb_emb = (float*)carve((size_t)BB * TT * EE * 4);
    float* y_emb  = (float*)carve((size_t)BB * TY * EE * 4);
    int*   xb_ids = (int*)carve((size_t)BB * TT * 4);
    bf16* enc_in_f = (bf16*)carve((size_t)BB * (EE + HHH) * 2);
    bf16* enc_in_b = (bf16*)carve((size_t)BB * (EE + HHH) * 2);
    bf16* dec_in   = (bf16*)carve((size_t)BB * (EE + 2 * HH2) * 2);
    float* gates_f = (float*)carve((size_t)BB * 4 * HHH * 4);
    float* gates_b = (float*)carve((size_t)BB * 4 * HHH * 4);
    float* gates_d = (float*)carve((size_t)BB * 4 * HH2 * 4);
    float* fh = (float*)carve((size_t)BB * HHH * 4);
    float* fc = (float*)carve((size_t)BB * HHH * 4);
    float* bh = (float*)carve((size_t)BB * HHH * 4);
    float* bc = (float*)carve((size_t)BB * HHH * 4);
    float* hiddensF = (float*)carve((size_t)BB * TT * HH2 * 4);
    bf16*  hiddensB = (bf16*) carve((size_t)BB * TT * HH2 * 2);
    float* cells    = (float*)carve((size_t)BB * TT * HH2 * 4);
    float* hproj    = (float*)carve((size_t)BB * TT * HH2 * 4);
    float* hx  = (float*)carve((size_t)BB * HH2 * 4);
    bf16*  hxB = (bf16*) carve((size_t)BB * HH2 * 2);
    float* cx  = (float*)carve((size_t)BB * HH2 * 4);
    float* ctx = (float*)carve((size_t)BB * HH2 * 4);
    float* shx = (float*)carve((size_t)BB * HH2 * 4);
    bf16*  preB   = (bf16*) carve((size_t)BB * TT * HH2 * 2);
    float* attlog = (float*)carve((size_t)BB * TT * HH2 * 4);
    bf16*  outhB  = (bf16*) carve((size_t)BB * TY * HH2 * 2);

    // ---- one-time weight prep (bf16, K-major) ----
    {
        long n;
        n = (long)(EE + HHH) * 4 * HHH;
        pack_cat2_T<<<(n + 255) / 256, 256, 0, stream>>>(fWih, EE, fWhh, HHH, fWcat, 4 * HHH);
        pack_cat2_T<<<(n + 255) / 256, 256, 0, stream>>>(bWih, EE, bWhh, HHH, bWcat, 4 * HHH);
        n = (long)(EE + 2 * HH2) * 4 * HH2;
        pack_cat2_T<<<(n + 255) / 256, 256, 0, stream>>>(dWih, EE + HH2, dWhh, HH2, dWcat, 4 * HH2);
        n = (long)HH2 * HH2;
        pack_cat2_T<<<(n + 255) / 256, 256, 0, stream>>>(sW, HH2, nullptr, 0, sWT, HH2);
        pack_cat2_T<<<(n + 255) / 256, 256, 0, stream>>>(hW, HH2, nullptr, 0, hWT, HH2);
        pack_cat2_T<<<(n + 255) / 256, 256, 0, stream>>>(aW, HH2, nullptr, 0, aWT, HH2);
        n = (long)HH2 * VV;
        pack_cat2_T<<<(n + 255) / 256, 256, 0, stream>>>(outW, HH2, nullptr, 0, outWT, VV);
        vec_add2<<<(4 * HHH + 255) / 256, 256, 0, stream>>>(fbih, fbhh, fbias, 4 * HHH);
        vec_add2<<<(4 * HHH + 255) / 256, 256, 0, stream>>>(bbih, bbhh, bbias, 4 * HHH);
        vec_add2<<<(4 * HH2 + 255) / 256, 256, 0, stream>>>(dbih, dbhh, dbias, 4 * HH2);
    }

    // ---- init states + embeddings ----
    copy_f32<<<(BB * HHH + 255) / 256, 256, 0, stream>>>(f_h0, fh, BB * HHH);
    copy_f32<<<(BB * HHH + 255) / 256, 256, 0, stream>>>(f_c0, fc, BB * HHH);
    copy_f32<<<(BB * HHH + 255) / 256, 256, 0, stream>>>(b_h0, bh, BB * HHH);
    copy_f32<<<(BB * HHH + 255) / 256, 256, 0, stream>>>(b_c0, bc, BB * HHH);

    gather_embed<<<BB * TT, EE, 0, stream>>>(x, embedW, x_emb, EE);
    build_rev_ids<<<BB, TT, 0, stream>>>(x, xmask, xb_ids, TT);
    gather_embed<<<BB * TT, EE, 0, stream>>>(xb_ids, embedW, xb_emb, EE);
    gather_embed<<<BB * TY, EE, 0, stream>>>(y, embedW, y_emb, EE);

    // ---- bidirectional encoder, T sequential steps ----
    for (int t = 0; t < TT; ++t) {
        concat3_cast<<<BB, 256, 0, stream>>>(x_emb + (size_t)t * EE, (long)TT * EE, EE,
                                             fh, HHH, HHH, nullptr, 0, 0, enc_in_f);
        concat3_cast<<<BB, 256, 0, stream>>>(xb_emb + (size_t)t * EE, (long)TT * EE, EE,
                                             bh, HHH, HHH, nullptr, 0, 0, enc_in_b);
        launch_gemm(enc_in_f, fWcat, fbias, gates_f, BB, 4 * HHH, EE + HHH, stream);
        launch_gemm(enc_in_b, bWcat, bbias, gates_b, BB, 4 * HHH, EE + HHH, stream);
        lstm_act<<<(BB * HHH + 255) / 256, 256, 0, stream>>>(
            gates_f, fc, fh, nullptr,
            hiddensF + (size_t)t * HH2,       (long)TT * HH2,
            hiddensB + (size_t)t * HH2,       (long)TT * HH2,
            cells    + (size_t)t * HH2,       (long)TT * HH2, HHH, BB);
        lstm_act<<<(BB * HHH + 255) / 256, 256, 0, stream>>>(
            gates_b, bc, bh, nullptr,
            hiddensF + (size_t)t * HH2 + HHH, (long)TT * HH2,
            hiddensB + (size_t)t * HH2 + HHH, (long)TT * HH2,
            cells    + (size_t)t * HH2 + HHH, (long)TT * HH2, HHH, BB);
    }

    // ---- decoder init + hproj ----
    mean_T<<<(BB * HH2 + 255) / 256, 256, 0, stream>>>(hiddensF, hx, TT, HH2);
    mean_T<<<(BB * HH2 + 255) / 256, 256, 0, stream>>>(cells, cx, TT, HH2);
    copy_f32<<<(BB * HH2 + 255) / 256, 256, 0, stream>>>(hx, ctx, BB * HH2);
    launch_gemm(hiddensB, hWT, hb, hproj, BB * TT, HH2, HH2, stream);

    // ---- decoder, TY sequential steps ----
    for (int t = 0; t < TY; ++t) {
        concat3_cast<<<BB, 256, 0, stream>>>(y_emb + (size_t)t * EE, (long)TY * EE, EE,
                                             ctx, HH2, HH2, hx, HH2, HH2, dec_in);
        launch_gemm(dec_in, dWcat, dbias, gates_d, BB, 4 * HH2, EE + 2 * HH2, stream);
        lstm_act<<<(BB * HH2 + 255) / 256, 256, 0, stream>>>(
            gates_d, cx, hx, hxB,
            out_h + (size_t)t * HH2, (long)TY * HH2,
            outhB + (size_t)t * HH2, (long)TY * HH2,
            nullptr, 0, HH2, BB);
        launch_gemm(hxB, sWT, sb, shx, BB, HH2, HH2, stream);
        add_bcast_cast<<<(BB * TT * HH2 + 255) / 256, 256, 0, stream>>>(shx, hproj, preB, TT, HH2);
        launch_gemm(preB, aWT, ab, attlog, BB * TT, HH2, HH2, stream);
        softmax_rows<<<BB * TT, 256, 0, stream>>>(attlog, HH2);
        ctx_reduce<<<(BB * HH2 + 255) / 256, 256, 0, stream>>>(hiddensF, attlog, ctx, TT, HH2);
    }

    // ---- output projection + log_softmax (written in-place into d_out) ----
    launch_gemm(outhB, outWT, outb, logits, BB * TY, VV, HH2, stream);
    log_softmax_rows<<<BB * TY, 256, 0, stream>>>(logits, VV);
}
